// RGCNPoolNet_89970974916671
// MI455X (gfx1250) — matrix operand order repaired and checked by reading, not compiled, not run on hardware
//
#include <hip/hip_runtime.h>
#include <hip/hip_bf16.h>
#include <math.h>

// ---------------- problem constants (match reference) ----------------
#define N_NODES   10000
#define E_CNT     80000
#define C_DIM     512
#define TWO_C     1024
#define NPROC     39999          // ceil(E*0.5)-1
#define NPAD      131072         // 2^17 >= E, bitonic size

// output layout (flat element offsets, 4B elements)
#define O_NEWX    0                         // N*C f32
#define O_SCORE   (N_NODES * C_DIM)         // E f32
#define O_NEI     (O_SCORE + E_CNT)         // 2E i32
#define O_CLUS    (O_NEI + 2 * E_CNT)       // N i32
#define O_NBATCH  (O_CLUS + N_NODES)        // N i32
#define O_NUMC    (O_NBATCH + N_NODES)      // 1 i32

// workspace layout (byte offsets)
#define WS_XB     0                                   // N*C bf16 (ushort)
#define WS_W2B    (WS_XB + N_NODES * C_DIM * 2)       // 1024*1024 bf16
#define WS_PART   (WS_W2B + TWO_C * TWO_C * 2)        // E f32
#define WS_KEYS   (WS_PART + E_CNT * 4)               // NPAD u64

typedef __attribute__((ext_vector_type(16))) __bf16 v16bf;
typedef __attribute__((ext_vector_type(8)))  float  v8f;

// ---------------- helpers ----------------
__device__ __forceinline__ unsigned short f32_to_bf16_rne(float f) {
    unsigned int u = __float_as_uint(f);
    unsigned int r = (u + 0x7FFFu + ((u >> 16) & 1u)) >> 16;
    return (unsigned short)r;
}

// ---------------- conversion / init kernels ----------------
__global__ void k_cvt_bf16(const float* __restrict__ in,
                           unsigned short* __restrict__ out, int n) {
    int i = blockIdx.x * blockDim.x + threadIdx.x;
    if (i < n) out[i] = f32_to_bf16_rne(in[i]);
}

__global__ void k_zero_f32(float* __restrict__ p, int n) {
    int i = blockIdx.x * blockDim.x + threadIdx.x;
    if (i < n) p[i] = 0.0f;
}

// ---------------- fused edge-MLP GEMM (bf16 WMMA) ----------------
// block: 256 threads (8 waves). M-tile = 32 edges staged in LDS, N = 1024 cols.
// Each wave: 1 M-subtile (16 rows) x 16 N-tiles, K = 1024 (32 WMMA per tile).
// Per-row score partials accumulate in registers across all N-tiles; the
// cross-lane reduction (4x shfl_xor) + atomic happens ONCE per wave.
#define MTILE 32
__global__ __launch_bounds__(256) void k_edge_mlp(
    const unsigned short* __restrict__ xb,     // [N, C] bf16
    const unsigned short* __restrict__ w2b,    // [2C, 2C] bf16 row-major (row j = output col)
    const int* __restrict__ src, const int* __restrict__ dst,
    const float* __restrict__ b2, const float* __restrict__ w1,
    float* __restrict__ partial)               // [E] f32, pre-zeroed
{
    __shared__ __align__(16) unsigned short As[MTILE * TWO_C];  // 64 KB

    const int tid   = threadIdx.x;
    const int mbase = blockIdx.x * MTILE;

    // ---- stage A: 32 edges x 1024 bf16 (concat of x[src], x[dst]) ----
    {
        const int grp = tid >> 3;     // 0..31 : edge row
        const int l8  = tid & 7;
        const int e   = mbase + grp;
        uint4* arow = (uint4*)(As + grp * TWO_C);   // 128 uint4 per row
        if (e < E_CNT) {
            const uint4* sr = (const uint4*)(xb + (size_t)src[e] * C_DIM); // 64 uint4
            const uint4* dr = (const uint4*)(xb + (size_t)dst[e] * C_DIM);
            #pragma unroll
            for (int i = l8; i < 64; i += 8) { arow[i] = sr[i]; arow[64 + i] = dr[i]; }
        } else {
            uint4 z = {0u, 0u, 0u, 0u};
            #pragma unroll
            for (int i = l8; i < 128; i += 8) arow[i] = z;
        }
    }
    __syncthreads();

    const int wave    = tid >> 5;        // 0..7
    const int lane    = tid & 31;
    const int l15     = lane & 15;
    const int kh      = lane >> 4;       // half-wave id (ISA 16-bit A/B K split)
    const int msub    = wave & 1;        // 2 M-subtiles of 16 rows
    const int nstream = wave >> 1;       // 4 N streams
    const unsigned short* aptr = As + (msub * 16 + l15) * TWO_C;

    float rowacc[8];
    #pragma unroll
    for (int i = 0; i < 8; ++i) rowacc[i] = 0.0f;

    for (int nt = nstream; nt < 64; nt += 4) {
        const int col = nt * 16 + l15;
        const unsigned short* bptr = w2b + (size_t)col * TWO_C + kh * 16;
        v8f c = {};
        #pragma unroll 4
        for (int kk = 0; kk < TWO_C; kk += 32) {
            v16bf a, b;
            // A 16x32 bf16 layout: lane<16: K 0-15 ; lane>=16: K 16-31
            uint4* ap = (uint4*)&a;
            ap[0] = *(const uint4*)(aptr + kk + kh * 16);
            ap[1] = *(const uint4*)(aptr + kk + kh * 16 + 8);
            // B 32x16 bf16: lane = column, lanes<16 hold K 0-15, lanes>=16 hold K 16-31
            uint4* bp = (uint4*)&b;
            const uint4* bg = (const uint4*)(bptr + kk);
            bp[0] = bg[0]; bp[1] = bg[1];
            c = __builtin_amdgcn_wmma_f32_16x16x32_bf16(
                    false, a, false, b, (short)0, c, false, false);
        }
        // fold this tile's 16 columns into per-lane row partials (no shuffles yet)
        const float bias = b2[col];
        const float w    = w1[col];
        #pragma unroll
        for (int i = 0; i < 8; ++i) {
            float v = c[i] + bias;
            v = v > 0.0f ? v : 0.0f;
            rowacc[i] = fmaf(v, w, rowacc[i]);
        }
    }

    // one cross-lane reduction + one atomic per row, per wave
    #pragma unroll
    for (int i = 0; i < 8; ++i) {
        float v = rowacc[i];
        #pragma unroll
        for (int off = 1; off < 16; off <<= 1) v += __shfl_xor(v, off, 32);
        if (l15 == 0) {
            const int row = msub * 16 + i + kh * 8;   // C layout: vgpr i -> M=i (+8 hi lanes)
            const int e = mbase + row;
            if (e < E_CNT) atomicAdd(&partial[e], v);
        }
    }
}

// ---------------- score finalize + sort-key build ----------------
__global__ void k_finalize(const float* __restrict__ partial,
                           const float* __restrict__ b1,
                           float* __restrict__ score_out,
                           unsigned long long* __restrict__ keys) {
    int i = blockIdx.x * blockDim.x + threadIdx.x;
    if (i >= NPAD) return;
    unsigned long long key = ~0ull;
    if (i < E_CNT) {
        float z = partial[i] + b1[0];
        float s = 1.0f / (1.0f + __expf(-z));
        score_out[i] = s;
        unsigned int u = __float_as_uint(s);
        u = (u >> 31) ? ~u : (u | 0x80000000u);   // monotone float->uint
        unsigned int k32 = ~u;                     // descending
        key = ((unsigned long long)k32 << 32) | (unsigned int)i;  // stable: tie-break by id
    }
    keys[i] = key;
}

__global__ void k_bitonic(unsigned long long* __restrict__ keys, int j, int k) {
    int i = blockIdx.x * blockDim.x + threadIdx.x;
    if (i >= NPAD) return;
    int ixj = i ^ j;
    if (ixj <= i) return;
    unsigned long long a = keys[i], b = keys[ixj];
    bool up = ((i & k) == 0);
    if ((a > b) == up) { keys[i] = b; keys[ixj] = a; }
}

// ---------------- greedy contraction (serial matching in LDS) ----------------
#define CCH 1024
__global__ __launch_bounds__(1024) void k_contract(
    const unsigned long long* __restrict__ keys,
    const int* __restrict__ src, const int* __restrict__ dst,
    const int* __restrict__ batch,
    int* __restrict__ out_cluster, int* __restrict__ out_nbatch,
    int* __restrict__ out_numc)
{
    __shared__ unsigned int maskb[(N_NODES + 31) / 32];
    __shared__ int clus[N_NODES];
    __shared__ int cs[CCH], cd[CCH];
    __shared__ int sh_i, sh_numc;

    const int tid = threadIdx.x;
    const int nt  = blockDim.x;

    for (int w = tid; w < (N_NODES + 31) / 32; w += nt) {
        int base = w * 32;
        int valid = N_NODES - base;
        unsigned int m = (valid >= 32) ? 0xFFFFFFFFu
                       : (valid <= 0 ? 0u : ((1u << valid) - 1u));
        maskb[w] = m;
    }
    for (int w = tid; w < N_NODES; w += nt) clus[w] = 0;
    if (tid == 0) sh_i = 0;
    __syncthreads();

    for (int base = 0; base < NPROC; base += CCH) {
        int cnt = NPROC - base; if (cnt > CCH) cnt = CCH;
        for (int t = tid; t < cnt; t += nt) {
            int e = (int)(keys[base + t] & 0xFFFFFFFFull);
            cs[t] = src[e]; cd[t] = dst[e];
        }
        __syncthreads();
        if (tid == 0) {
            int i = sh_i;
            for (int t = 0; t < cnt; ++t) {
                int s = cs[t], d = cd[t];
                unsigned int bs = (maskb[s >> 5] >> (s & 31)) & 1u;
                unsigned int bd = (maskb[d >> 5] >> (d & 31)) & 1u;
                if (bs & bd) {
                    clus[s] = i; clus[d] = i;
                    maskb[s >> 5] &= ~(1u << (s & 31));
                    maskb[d >> 5] &= ~(1u << (d & 31));
                    ++i;
                }
            }
            sh_i = i;
        }
        __syncthreads();
    }

    if (tid == 0) {   // singleton clusters i..i+j-1 in node order (cumsum of mask)
        int r = sh_i;
        for (int n = 0; n < N_NODES; ++n)
            if ((maskb[n >> 5] >> (n & 31)) & 1u) clus[n] = r++;
        sh_numc = r;
    }
    __syncthreads();

    for (int n = tid; n < N_NODES; n += nt) out_cluster[n] = clus[n];
    for (int n = tid; n < N_NODES; n += nt) out_nbatch[n] = 0;
    __syncthreads();
    for (int n = tid; n < N_NODES; n += nt) out_nbatch[clus[n]] = batch[n];
    if (tid == 0) out_numc[0] = sh_numc;
}

// ---------------- scatters ----------------
__global__ void k_newx_scatter(const float* __restrict__ x,
                               const int* __restrict__ cluster,
                               float* __restrict__ newx) {
    int i = blockIdx.x * blockDim.x + threadIdx.x;
    if (i >= N_NODES * C_DIM) return;
    int n = i >> 9;            // /512
    int c = i & (C_DIM - 1);
    atomicAdd(&newx[(size_t)cluster[n] * C_DIM + c], x[i]);
}

__global__ void k_new_edges(const int* __restrict__ edge_index,
                            const int* __restrict__ cluster,
                            int* __restrict__ nei) {
    int i = blockIdx.x * blockDim.x + threadIdx.x;
    if (i >= 2 * E_CNT) return;
    nei[i] = cluster[edge_index[i]];
}

// ---------------- driver ----------------
extern "C" void kernel_launch(void* const* d_in, const int* in_sizes, int n_in,
                              void* d_out, int out_size, void* d_ws, size_t ws_size,
                              hipStream_t stream) {
    const float* x     = (const float*)d_in[0];
    const int*   ei    = (const int*)d_in[1];
    const int*   batch = (const int*)d_in[2];
    const float* W2    = (const float*)d_in[3];
    const float* b2    = (const float*)d_in[4];
    const float* W1    = (const float*)d_in[5];
    const float* b1    = (const float*)d_in[6];
    const int* src = ei;
    const int* dst = ei + E_CNT;

    char* ws = (char*)d_ws;
    unsigned short* xb   = (unsigned short*)(ws + WS_XB);
    unsigned short* w2b  = (unsigned short*)(ws + WS_W2B);
    float*          part = (float*)(ws + WS_PART);
    unsigned long long* keys = (unsigned long long*)(ws + WS_KEYS);

    float* out_f   = (float*)d_out;
    int*   out_i   = (int*)d_out;
    float* newx    = out_f + O_NEWX;
    float* score   = out_f + O_SCORE;
    int*   nei     = out_i + O_NEI;
    int*   clus    = out_i + O_CLUS;
    int*   nbatch  = out_i + O_NBATCH;
    int*   numc    = out_i + O_NUMC;

    const int T = 256;

    // 1) bf16 conversions
    k_cvt_bf16<<<(N_NODES * C_DIM + T - 1) / T, T, 0, stream>>>(x, xb, N_NODES * C_DIM);
    k_cvt_bf16<<<(TWO_C * TWO_C + T - 1) / T, T, 0, stream>>>(W2, w2b, TWO_C * TWO_C);

    // 2) zero partial scores and new_x output
    k_zero_f32<<<(E_CNT + T - 1) / T, T, 0, stream>>>(part, E_CNT);
    k_zero_f32<<<(N_NODES * C_DIM + T - 1) / T, T, 0, stream>>>(newx, N_NODES * C_DIM);

    // 3) fused edge MLP GEMM (WMMA bf16)
    k_edge_mlp<<<(E_CNT + MTILE - 1) / MTILE, 256, 0, stream>>>(
        xb, w2b, src, dst, b2, W1, part);

    // 4) sigmoid + sort keys
    k_finalize<<<NPAD / T, T, 0, stream>>>(part, b1, score, keys);

    // 5) bitonic sort (ascending 64-bit keys == stable descending by score)
    for (int k = 2; k <= NPAD; k <<= 1)
        for (int j = k >> 1; j > 0; j >>= 1)
            k_bitonic<<<NPAD / T, T, 0, stream>>>(keys, j, k);

    // 6) greedy contraction (single workgroup, LDS state)
    k_contract<<<1, 1024, 0, stream>>>(keys, src, dst, batch, clus, nbatch, numc);

    // 7) segment_sum of x into new_x
    k_newx_scatter<<<(N_NODES * C_DIM + T - 1) / T, T, 0, stream>>>(x, clus, newx);

    // 8) relabel edges
    k_new_edges<<<(2 * E_CNT + T - 1) / T, T, 0, stream>>>(ei, clus, nei);
}
